// GPT2_26938034880887
// MI455X (gfx1250) — matrix operand order repaired and checked by reading, not compiled, not run on hardware
//
#include <hip/hip_runtime.h>

// ---------------------------------------------------------------------------
// GPT-2 small forward for gfx1250 (MI455X). bf16 WMMA everywhere, fp32 accum.
// Memory-bound on fp32 weight streaming (~0.5 GB @ 23.3 TB/s); weights are
// converted fp32->bf16 during LDS staging, activations live in bf16.
// DPP-based 16-lane softmax reductions (no ds_bpermute), async global->LDS
// staging (ASYNCcnt) for the pure-copy tiles.
// ---------------------------------------------------------------------------

typedef unsigned short u16;
typedef unsigned int   u32;
typedef __attribute__((ext_vector_type(16))) __bf16 bf16x16;
typedef __attribute__((ext_vector_type(8)))  float  f32x8;

struct alignas(16) U4 { u32 x, y, z, w; };
union Frag { U4 u[2]; bf16x16 v; };

constexpr int kL  = 12;
constexpr int kD  = 768;
constexpr int kNH = 12;
constexpr int kDH = 64;
constexpr int kDM = 3072;
constexpr int kT  = 1024;
constexpr int kV  = 50257;
constexpr int kB  = 2;
constexpr int kMT = kB * kT;   // 2048 token rows

#if defined(__gfx1250__) && __has_builtin(__builtin_amdgcn_global_load_async_to_lds_b128)
#define USE_ASYNC_LDS 1
#else
#define USE_ASYNC_LDS 0
#endif

#if USE_ASYNC_LDS
// Signature (from clang diagnostic): (v4i AS1*, v4i AS3*, imm offset, imm cpol)
typedef int v4i_ __attribute__((__vector_size__(16)));
typedef __attribute__((address_space(1))) v4i_ gv4i_;
typedef __attribute__((address_space(3))) v4i_ lv4i_;
static __device__ __forceinline__ void async_copy16(const void* g, void* l) {
  // GLOBAL_LOAD_ASYNC_TO_LDS_B128: 16B/lane, tracked by ASYNCcnt, no VGPR data.
  __builtin_amdgcn_global_load_async_to_lds_b128((gv4i_*)g, (lv4i_*)l, 0, 0);
}
#define ASYNC_FENCE() asm volatile("s_wait_asynccnt 0x0" ::: "memory")
#endif

static __device__ __forceinline__ u16 f2bf(float x) {
  union { float f; u32 u; } c; c.f = x;
  return (u16)((c.u + 0x7FFFu + ((c.u >> 16) & 1u)) >> 16);  // RNE
}

// --- VALU-only reductions over a 16-lane DPP row (no LDS traffic) ----------
template <int CTRL>
static __device__ __forceinline__ float dppmov(float x) {
  union { float f; int i; } a, r;
  a.f = x;
  r.i = __builtin_amdgcn_update_dpp(0, a.i, CTRL, 0xF, 0xF, true);
  return r.f;
}
static __device__ __forceinline__ float red16_max(float x) {
  x = fmaxf(x, dppmov<0xB1>(x));    // quad_perm [1,0,3,2]  == xor1
  x = fmaxf(x, dppmov<0x4E>(x));    // quad_perm [2,3,0,1]  == xor2
  x = fmaxf(x, dppmov<0x141>(x));   // row_half_mirror      == xor4 (quads uniform)
  x = fmaxf(x, dppmov<0x140>(x));   // row_mirror           == xor8 (octs uniform)
  return x;
}
static __device__ __forceinline__ float red16_sum(float x) {
  x += dppmov<0xB1>(x);
  x += dppmov<0x4E>(x);
  x += dppmov<0x141>(x);
  x += dppmov<0x140>(x);
  return x;
}

// ---------------------------------------------------------------------------
// Embedding: x[b,t,:] = emb[id[b,t],:] + pos[t,:]   (fp32 residual stream)
// ---------------------------------------------------------------------------
__global__ __launch_bounds__(256) void embed_kernel(
    const int* __restrict__ ids, const float* __restrict__ emb,
    const float* __restrict__ pos, float* __restrict__ x) {
  const int row = blockIdx.x;           // b*T + t
  const int t   = row & (kT - 1);
  const int id  = ids[row];
  const float* e = emb + (size_t)id * kD;
  const float* p = pos + (size_t)t * kD;
  float* xr = x + (size_t)row * kD;
  for (int i = threadIdx.x; i < kD; i += 256) xr[i] = e[i] + p[i];
}

// ---------------------------------------------------------------------------
// LayerNorm: fp32 in -> bf16 out. One block (256 thr) per row of 768.
// ---------------------------------------------------------------------------
__global__ __launch_bounds__(256) void ln_kernel(
    const float* __restrict__ x, const float* __restrict__ g,
    const float* __restrict__ b, u16* __restrict__ out) {
  __shared__ float red[256];
  const int row = blockIdx.x;
  const int t   = threadIdx.x;
  const float* xr = x + (size_t)row * kD;
  float v0 = xr[t], v1 = xr[t + 256], v2 = xr[t + 512];
  red[t] = v0 + v1 + v2;
  __syncthreads();
  for (int off = 128; off > 0; off >>= 1) {
    if (t < off) red[t] += red[t + off];
    __syncthreads();
  }
  const float mu = red[0] * (1.0f / kD);
  __syncthreads();
  const float d0 = v0 - mu, d1 = v1 - mu, d2 = v2 - mu;
  red[t] = d0 * d0 + d1 * d1 + d2 * d2;
  __syncthreads();
  for (int off = 128; off > 0; off >>= 1) {
    if (t < off) red[t] += red[t + off];
    __syncthreads();
  }
  const float rstd = rsqrtf(red[0] * (1.0f / kD) + 1e-5f);
  u16* orow = out + (size_t)row * kD;
  orow[t]       = f2bf(d0 * rstd * g[t]       + b[t]);
  orow[t + 256] = f2bf(d1 * rstd * g[t + 256] + b[t + 256]);
  orow[t + 512] = f2bf(d2 * rstd * g[t + 512] + b[t + 512]);
}

// ---------------------------------------------------------------------------
// Tiled WMMA GEMM:  C[M,N] = act( A[M,K](bf16) * W[K,N](fp32) + bias )
// Block tile 128x128, K-step 32, 8 waves (4 along M x 2 along N),
// wave tile 32x64 -> 8 x v_wmma_f32_16x16x32_bf16 per K-step.
// A tile: pure bf16 copy -> async global->LDS.
// W tile: fp32->bf16 convert + transpose through VGPRs into LDS ([n][k]).
// ---------------------------------------------------------------------------
template <bool GELU, bool RESID, bool OUTF, bool OUTB>
__global__ __launch_bounds__(256) void gemm_bf16(
    const u16* __restrict__ A, int lda,
    const float* __restrict__ W, int N, int K,
    const float* __restrict__ bias,
    const float* __restrict__ resid,   // fp32 [M,N], row stride N
    float* __restrict__ outf, u16* __restrict__ outb) {
  __shared__ u16 sA[128][48];   // [m][k], row = 96B (16B aligned)
  __shared__ u16 sB[128][48];   // [n][k]

  const int tid  = threadIdx.x;
  const int lane = tid & 31, wave = tid >> 5;
  const int wm   = wave & 3,  wn   = wave >> 2;
  const int half = lane >> 4, l16  = lane & 15;
  const int mBase = blockIdx.y * 128;
  const int nBase = blockIdx.x * 128;

  f32x8 acc[2][4];
#pragma unroll
  for (int i = 0; i < 2; ++i)
#pragma unroll
    for (int j = 0; j < 4; ++j)
#pragma unroll
      for (int e = 0; e < 8; ++e) acc[i][j][e] = 0.0f;

  const int am   = tid >> 1;        // A staging: row 0..127
  const int aseg = tid & 1;         //            16-elem k segment
  const int bk   = (tid & 15) * 2;  // W staging: even k pair
  const int bn   = (tid >> 4) * 8;  //            8-col group

  for (int k0 = 0; k0 < K; k0 += 32) {
    // ---- fetch W tile into regs (before barrier, overlaps prior compute) --
    const float* w0 = W + (size_t)(k0 + bk) * N + nBase + bn;
    const float* w1 = w0 + N;
    float r0[8], r1[8];
#pragma unroll
    for (int i = 0; i < 8; ++i) {
      const bool ok = (nBase + bn + i) < N;   // ragged-N (LM head) guard
      r0[i] = ok ? w0[i] : 0.0f;
      r1[i] = ok ? w1[i] : 0.0f;
    }
    if (k0 + 32 < K) __builtin_prefetch(w0 + (size_t)32 * N, 0, 1);

    const u16* ga = A + (size_t)(mBase + am) * lda + k0 + aseg * 16;
#if !USE_ASYNC_LDS
    U4 a0 = *(const U4*)ga;
    U4 a1 = *((const U4*)ga + 1);
#endif

    __syncthreads();                 // prior iteration's LDS reads done
#if USE_ASYNC_LDS
    // A tile: async DMA into LDS (must start only after the barrier).
    async_copy16(ga,     &sA[am][aseg * 16]);
    async_copy16(ga + 8, &sA[am][aseg * 16 + 8]);
#else
    *(U4*)&sA[am][aseg * 16]       = a0;
    *((U4*)&sA[am][aseg * 16] + 1) = a1;
#endif
#pragma unroll
    for (int i = 0; i < 8; ++i) {
      u32 pk = (u32)f2bf(r0[i]) | ((u32)f2bf(r1[i]) << 16);
      *(u32*)&sB[bn + i][bk] = pk;   // transposed: [n][k]
    }
#if USE_ASYNC_LDS
    ASYNC_FENCE();                   // my async writes landed in LDS
#endif
    __syncthreads();

    // ---- fragments + WMMA ----
    Frag a[2];
#pragma unroll
    for (int mi = 0; mi < 2; ++mi) {
      const u16* p = &sA[wm * 32 + mi * 16 + l16][0];
      a[mi].u[0] = *(const U4*)(p + half * 16);       // A: K = half*16 + 0..7
      a[mi].u[1] = *(const U4*)(p + half * 16 + 8);   //    K = half*16+8 ..
    }
#pragma unroll
    for (int ni = 0; ni < 4; ++ni) {
      Frag bfr;
      const u16* p = &sB[wn * 64 + ni * 16 + l16][0];
      bfr.u[0] = *(const U4*)(p + half * 16);         // B: K = half*16 + 0..15
      bfr.u[1] = *(const U4*)(p + half * 16 + 8);
#pragma unroll
      for (int mi = 0; mi < 2; ++mi)
        acc[mi][ni] = __builtin_amdgcn_wmma_f32_16x16x32_bf16(
            false, a[mi].v, false, bfr.v, (short)0, acc[mi][ni], false, false);
    }
  }

  // ---- epilogue: bias (+GELU) (+residual), write fp32 and/or bf16 ----
#pragma unroll
  for (int ni = 0; ni < 4; ++ni) {
    const int col = nBase + wn * 64 + ni * 16 + l16;
    if (col < N) {
      const float bcol = bias[col];
#pragma unroll
      for (int mi = 0; mi < 2; ++mi) {
#pragma unroll
        for (int r = 0; r < 8; ++r) {
          const int row = mBase + wm * 32 + mi * 16 + half * 8 + r;
          float v = acc[mi][ni][r] + bcol;
          if (GELU) v = 0.5f * v * (1.0f + erff(v * 0.70710678118654752f));
          if (RESID) v += resid[(size_t)row * N + col];
          if (OUTF) outf[(size_t)row * N + col] = v;
          if (OUTB) outb[(size_t)row * N + col] = f2bf(v);
        }
      }
    }
  }
}

// ---------------------------------------------------------------------------
// Flash attention, causal, scale = 1/8. One block per (b*NH + h, 128 q rows).
// 8 waves, each owns 16 q rows. Q fragments live in registers for the whole
// key sweep; keys processed in 64-wide chunks with online softmax.
// K tile is a pure copy -> async global->LDS; V needs a register transpose.
// ---------------------------------------------------------------------------
__global__ __launch_bounds__(256) void attn_kernel(
    const u16* __restrict__ q, const u16* __restrict__ k,
    const u16* __restrict__ v, u16* __restrict__ z) {
  __shared__ u16 sK [64][72];       // [key][dh]   (B-frag of Q*K^T)
  __shared__ u16 sVT[64][72];       // [dh][key]   (B-frag of P*V)
  __shared__ u16 sP [8][16][72];    // per-wave P transpose buffer

  const int tid  = threadIdx.x;
  const int lane = tid & 31, wave = tid >> 5;
  const int half = lane >> 4, l16  = lane & 15;
  const int bh = blockIdx.x;
  const int b  = bh / kNH, h = bh % kNH;
  const int qb = blockIdx.y * 128;
  const size_t rowOff = (size_t)b * kT;
  const int colOff = h * kDH;

  // Q fragments (A-layout) straight from global, kept in registers.
  Frag qf[2];
  {
    const u16* qp = q + (rowOff + qb + wave * 16 + l16) * kD + colOff;
    qf[0].u[0] = *(const U4*)(qp + half * 16);
    qf[0].u[1] = *(const U4*)(qp + half * 16 + 8);
    qf[1].u[0] = *(const U4*)(qp + 32 + half * 16);
    qf[1].u[1] = *(const U4*)(qp + 32 + half * 16 + 8);
  }

  float mrow[8], lrow[8];
  f32x8 o[4];
#pragma unroll
  for (int r = 0; r < 8; ++r) { mrow[r] = -3.0e38f; lrow[r] = 0.0f; }
#pragma unroll
  for (int f = 0; f < 4; ++f)
#pragma unroll
    for (int e = 0; e < 8; ++e) o[f][e] = 0.0f;

  const int nChunks = qb / 64 + 2;          // keys 0 .. qb+127
  for (int c = 0; c < nChunks; ++c) {
    const int kb = c * 64;
    __syncthreads();
    {   // stage K chunk: [key][dh]  (pure copy -> async DMA)
      const int key = tid >> 2, seg = tid & 3;
      const u16* gk = k + (rowOff + kb + key) * kD + colOff + seg * 16;
#if USE_ASYNC_LDS
      async_copy16(gk,     &sK[key][seg * 16]);
      async_copy16(gk + 8, &sK[key][seg * 16 + 8]);
#else
      *(U4*)&sK[key][seg * 16]       = *(const U4*)gk;
      *((U4*)&sK[key][seg * 16] + 1) = *((const U4*)gk + 1);
#endif
    }
    {   // stage V chunk transposed: [dh][key]  (register transpose)
      const int key0 = (tid & 31) * 2, dh0 = (tid >> 5) * 8;
      const u16* v0p = v + (rowOff + kb + key0) * kD + colOff + dh0;
      union { U4 u; u16 s[8]; } va, vb;
      va.u = *(const U4*)v0p;
      vb.u = *(const U4*)(v0p + kD);
#pragma unroll
      for (int i = 0; i < 8; ++i)
        *(u32*)&sVT[dh0 + i][key0] = (u32)va.s[i] | ((u32)vb.s[i] << 16);
    }
#if USE_ASYNC_LDS
    ASYNC_FENCE();
#endif
    __syncthreads();

    // S = Q * K^T  (16 q rows x 64 keys per wave)
    f32x8 sacc[4];
#pragma unroll
    for (int ni = 0; ni < 4; ++ni)
#pragma unroll
      for (int e = 0; e < 8; ++e) sacc[ni][e] = 0.0f;
#pragma unroll
    for (int ni = 0; ni < 4; ++ni) {
      Frag kb0, kb1;
      const u16* kp = &sK[ni * 16 + l16][0];
      kb0.u[0] = *(const U4*)(kp + half * 16);         // dh 0..31
      kb0.u[1] = *(const U4*)(kp + half * 16 + 8);
      kb1.u[0] = *(const U4*)(kp + 32 + half * 16);    // dh 32..63
      kb1.u[1] = *(const U4*)(kp + 32 + half * 16 + 8);
      sacc[ni] = __builtin_amdgcn_wmma_f32_16x16x32_bf16(
          false, qf[0].v, false, kb0.v, (short)0, sacc[ni], false, false);
      sacc[ni] = __builtin_amdgcn_wmma_f32_16x16x32_bf16(
          false, qf[1].v, false, kb1.v, (short)0, sacc[ni], false, false);
    }

    // scale + causal mask (matches reference: scale, then -1e5 where masked)
    const int qrow0 = qb + wave * 16 + half * 8;
#pragma unroll
    for (int ni = 0; ni < 4; ++ni) {
      const int key = kb + ni * 16 + l16;
#pragma unroll
      for (int r = 0; r < 8; ++r) {
        float s = sacc[ni][r] * 0.125f;
        if (key > qrow0 + r) s = -100000.0f;
        sacc[ni][r] = s;
      }
    }

    // online softmax; a row lives on 16 lanes -> DPP row reductions (VALU only)
    float alpha[8];
#pragma unroll
    for (int r = 0; r < 8; ++r) {
      float mx = fmaxf(fmaxf(sacc[0][r], sacc[1][r]),
                       fmaxf(sacc[2][r], sacc[3][r]));
      mx = red16_max(mx);
      const float mn = fmaxf(mrow[r], mx);
      alpha[r] = __expf(mrow[r] - mn);
      mrow[r] = mn;
    }
#pragma unroll
    for (int r = 0; r < 8; ++r) {
      float rs = 0.0f;
#pragma unroll
      for (int ni = 0; ni < 4; ++ni) {
        float p = __expf(sacc[ni][r] - mrow[r]);
        sacc[ni][r] = p;
        rs += p;
      }
      rs = red16_sum(rs);
      lrow[r] = lrow[r] * alpha[r] + rs;
    }
#pragma unroll
    for (int f = 0; f < 4; ++f)
#pragma unroll
      for (int r = 0; r < 8; ++r) o[f][r] *= alpha[r];

    // P: accumulator layout -> A-fragment layout via per-wave LDS
#pragma unroll
    for (int ni = 0; ni < 4; ++ni)
#pragma unroll
      for (int r = 0; r < 8; ++r)
        sP[wave][half * 8 + r][ni * 16 + l16] = f2bf(sacc[ni][r]);
    asm volatile("s_wait_dscnt 0x0" ::: "memory");   // wave-local LDS RAW
    Frag pA0, pA1;
    {
      const u16* pp = &sP[wave][l16][0];
      pA0.u[0] = *(const U4*)(pp + half * 16);         // keys 0..31
      pA0.u[1] = *(const U4*)(pp + half * 16 + 8);
      pA1.u[0] = *(const U4*)(pp + 32 + half * 16);    // keys 32..63
      pA1.u[1] = *(const U4*)(pp + 32 + half * 16 + 8);
    }
#pragma unroll
    for (int f = 0; f < 4; ++f) {
      Frag vb0, vb1;
      const u16* vp = &sVT[f * 16 + l16][0];
      vb0.u[0] = *(const U4*)(vp + half * 16);         // keys 0..31
      vb0.u[1] = *(const U4*)(vp + half * 16 + 8);
      vb1.u[0] = *(const U4*)(vp + 32 + half * 16);    // keys 32..63
      vb1.u[1] = *(const U4*)(vp + 32 + half * 16 + 8);
      o[f] = __builtin_amdgcn_wmma_f32_16x16x32_bf16(
          false, pA0.v, false, vb0.v, (short)0, o[f], false, false);
      o[f] = __builtin_amdgcn_wmma_f32_16x16x32_bf16(
          false, pA1.v, false, vb1.v, (short)0, o[f], false, false);
    }
  }

  // normalize and write Z (bf16, [b*T+t, h*64+dh])
#pragma unroll
  for (int r = 0; r < 8; ++r) {
    const float inv = 1.0f / lrow[r];
    const size_t row = rowOff + qb + wave * 16 + half * 8 + r;
#pragma unroll
    for (int f = 0; f < 4; ++f)
      z[row * kD + colOff + f * 16 + l16] = f2bf(o[f][r] * inv);
  }
}

// ---------------------------------------------------------------------------
// Host orchestration
// ---------------------------------------------------------------------------
extern "C" void kernel_launch(void* const* d_in, const int* in_sizes, int n_in,
                              void* d_out, int out_size, void* d_ws, size_t ws_size,
                              hipStream_t stream) {
  (void)in_sizes; (void)n_in; (void)out_size; (void)ws_size;

  const int*   ids  = (const int*)  d_in[0];
  const float* emb  = (const float*)d_in[1];
  const float* pos  = (const float*)d_in[2];
  const float* Wq   = (const float*)d_in[3];
  const float* bq   = (const float*)d_in[4];
  const float* Wk   = (const float*)d_in[5];
  const float* bk   = (const float*)d_in[6];
  const float* Wv   = (const float*)d_in[7];
  const float* bv   = (const float*)d_in[8];
  const float* Wo   = (const float*)d_in[9];
  const float* bo   = (const float*)d_in[10];
  const float* ln1g = (const float*)d_in[11];
  const float* ln1b = (const float*)d_in[12];
  const float* ln2g = (const float*)d_in[13];
  const float* ln2b = (const float*)d_in[14];
  const float* W1   = (const float*)d_in[15];
  const float* b1   = (const float*)d_in[16];
  const float* W2   = (const float*)d_in[17];
  const float* b2   = (const float*)d_in[18];
  const float* lnfg = (const float*)d_in[19];
  const float* lnfb = (const float*)d_in[20];
  const float* Wu   = (const float*)d_in[21];
  const float* bu   = (const float*)d_in[22];
  float* out = (float*)d_out;

  // workspace carve-up (~35 MB)
  char* ws = (char*)d_ws;
  size_t off = 0;
  auto carve = [&](size_t bytes) {
    void* p = ws + off;
    off += (bytes + 255) & ~(size_t)255;
    return p;
  };
  float* x   = (float*)carve((size_t)kMT * kD * 4);   // fp32 residual stream
  u16*  hb   = (u16*)carve((size_t)kMT * kD * 2);     // LN output (bf16)
  u16*  qbuf = (u16*)carve((size_t)kMT * kD * 2);
  u16*  kbuf = (u16*)carve((size_t)kMT * kD * 2);
  u16*  vbuf = (u16*)carve((size_t)kMT * kD * 2);
  u16*  zbuf = (u16*)carve((size_t)kMT * kD * 2);
  u16*  mbuf = (u16*)carve((size_t)kMT * kDM * 2);    // MLP mid (bf16)

  const dim3 blk(256);
  const dim3 gD (kD  / 128, kMT / 128);   // (6,16)
  const dim3 gDM(kDM / 128, kMT / 128);   // (24,16)
  const dim3 gV ((kV + 127) / 128, kMT / 128);

  embed_kernel<<<kMT, blk, 0, stream>>>(ids, emb, pos, x);

  for (int l = 0; l < kL; ++l) {
    const size_t wDD = (size_t)l * kD * kD;
    ln_kernel<<<kMT, blk, 0, stream>>>(x, ln1g + l * kD, ln1b + l * kD, hb);
    gemm_bf16<false, false, false, true><<<gD, blk, 0, stream>>>(
        hb, kD, Wq + wDD, kD, kD, bq + l * kD, nullptr, nullptr, qbuf);
    gemm_bf16<false, false, false, true><<<gD, blk, 0, stream>>>(
        hb, kD, Wk + wDD, kD, kD, bk + l * kD, nullptr, nullptr, kbuf);
    gemm_bf16<false, false, false, true><<<gD, blk, 0, stream>>>(
        hb, kD, Wv + wDD, kD, kD, bv + l * kD, nullptr, nullptr, vbuf);
    attn_kernel<<<dim3(kB * kNH, kT / 128), blk, 0, stream>>>(qbuf, kbuf, vbuf, zbuf);
    gemm_bf16<false, true, true, false><<<gD, blk, 0, stream>>>(
        zbuf, kD, Wo + wDD, kD, kD, bo + l * kD, x, x, nullptr);        // x += z@Wo+bo
    ln_kernel<<<kMT, blk, 0, stream>>>(x, ln2g + l * kD, ln2b + l * kD, hb);
    gemm_bf16<true, false, false, true><<<gDM, blk, 0, stream>>>(
        hb, kD, W1 + (size_t)l * kD * kDM, kDM, kD, b1 + (size_t)l * kDM,
        nullptr, nullptr, mbuf);                                        // GELU
    gemm_bf16<false, true, true, false><<<gD, blk, 0, stream>>>(
        mbuf, kDM, W2 + (size_t)l * kDM * kD, kD, kDM, b2 + l * kD, x, x, nullptr);
  }

  ln_kernel<<<kMT, blk, 0, stream>>>(x, lnfg, lnfb, hb);
  gemm_bf16<false, false, true, false><<<gV, blk, 0, stream>>>(
      hb, kD, Wu, kV, kD, bu, nullptr, out, nullptr);                   // logits
}
